// TaylorLayer_72060961292646
// MI455X (gfx1250) — matrix operand order repaired and checked by reading, not compiled, not run on hardware
//
#include <hip/hip_runtime.h>

// ---------------------------------------------------------------------------
// TaylorLayer: y[n,o] = sum_{i,k} x[n,i]^k * coeffs[o,i,k] + bias[o]
// == GEMM: P[N=8192, K=4096] x C^T[K=4096, O=1024],
//   P[n, i*4+k] = x[n,i]^k, C[o, i*4+k] = coeffs[o,i,k] (contiguous).
// Compute-bound (860 FLOP/B) -> bf16 WMMA pipe with hi/lo error-compensated
// split (3 WMMAs: hi*hi + hi*lo + lo*hi) for ~fp32 accuracy at bf16 rates.
// ---------------------------------------------------------------------------

typedef __attribute__((ext_vector_type(16))) __bf16 v16bf;
typedef __attribute__((ext_vector_type(8)))  float  v8f;

constexpr int IN_DIM   = 1024;
constexpr int OUT_DIMC = 1024;
constexpr int KDIM     = IN_DIM * 4;     // 4096
constexpr int BLK_M    = 128;
constexpr int BLK_N    = 64;
constexpr int KT       = 32;             // bf16 WMMA K per step (8 x-values)
constexpr int KSTEPS   = KDIM / KT;      // 128
constexpr int LSTR     = KT + 8;         // padded LDS row stride (ushorts)

struct Smem {
  unsigned short Ah[2][BLK_M * LSTR];
  unsigned short Al[2][BLK_M * LSTR];
  unsigned short Bh[2][BLK_N * LSTR];
  unsigned short Bl[2][BLK_N * LSTR];
};

__device__ __forceinline__ unsigned short bf16_rne(float f) {
  unsigned int u = __float_as_uint(f);
  return (unsigned short)((u + 0x7FFFu + ((u >> 16) & 1u)) >> 16);
}
__device__ __forceinline__ unsigned short bf16_split(float f, float& rem) {
  unsigned int u = __float_as_uint(f);
  unsigned int h = (u + 0x7FFFu + ((u >> 16) & 1u)) >> 16;
  rem = f - __uint_as_float(h << 16);
  return (unsigned short)h;
}

__device__ __forceinline__ v8f wmma_bf16(v16bf a, v16bf b, v8f c) {
  // (neg_a, A, neg_b, B, c_mod, C, reuse_a, reuse_b)
  return __builtin_amdgcn_wmma_f32_16x16x32_bf16(false, a, false, b,
                                                 (short)0, c, false, false);
}

union Pack16 { unsigned short s[16]; uint4 q[2]; };
union Pack8  { unsigned short s[8];  uint4 q;    };
union FragU  { uint4 q[2]; v16bf v; };

__global__ __launch_bounds__(256)
void taylor_wmma_kernel(const float* __restrict__ x,
                        const float* __restrict__ coeffs,
                        const float* __restrict__ bias,
                        float* __restrict__ out) {
  __shared__ __align__(16) Smem sm;

  const int tid  = threadIdx.x;
  const int lane = tid & 31;
  const int wave = tid >> 5;       // 0..7
  const int lr   = lane & 15;      // row/col within 16-tile
  const int lh   = lane >> 4;      // half-wave

  const int mBase = blockIdx.x * BLK_M;
  const int oBase = blockIdx.y * BLK_N;

  // Global-load thread mapping (coalesced float4s)
  const int aRow = tid >> 1;       // 0..127 rows of A tile
  const int aSeg = tid & 1;        // which half of 8 x-values
  const int bRow = tid >> 2;       // 0..63 rows of B tile
  const int bSeg = tid & 3;        // which 8-float segment of 32

  const float* xPtr = x      + (size_t)(mBase + aRow) * IN_DIM + aSeg * 4;
  const float* cPtr = coeffs + (size_t)(oBase + bRow) * KDIM   + bSeg * 8;

  float4 aReg;
  float4 bReg0, bReg1;

  v8f acc[2][2];
  v8f zero = {};
  acc[0][0] = zero; acc[0][1] = zero; acc[1][0] = zero; acc[1][1] = zero;

  auto loadG = [&](int ks) {
    aReg  = *(const float4*)(xPtr + (size_t)ks * 8);     // 8 x-values / row
    bReg0 = *(const float4*)(cPtr + (size_t)ks * KT);
    bReg1 = *(const float4*)(cPtr + (size_t)ks * KT + 4);
  };

  auto storeS = [&](int st) {
    // --- A tile: expand each x into powers {1, x, x^2, x^3}, split hi/lo ---
    Pack16 hiA, loA;
    float xv[4] = {aReg.x, aReg.y, aReg.z, aReg.w};
#pragma unroll
    for (int j = 0; j < 4; ++j) {
      float v1 = xv[j];
      float v2 = v1 * v1;
      float v3 = v2 * v1;
      float p[4] = {1.0f, v1, v2, v3};
#pragma unroll
      for (int k = 0; k < 4; ++k) {
        float rem;
        hiA.s[j * 4 + k] = bf16_split(p[k], rem);
        loA.s[j * 4 + k] = bf16_rne(rem);
      }
    }
    {
      unsigned short* dH = &sm.Ah[st][aRow * LSTR + aSeg * 16];
      unsigned short* dL = &sm.Al[st][aRow * LSTR + aSeg * 16];
      ((uint4*)dH)[0] = hiA.q[0]; ((uint4*)dH)[1] = hiA.q[1];
      ((uint4*)dL)[0] = loA.q[0]; ((uint4*)dL)[1] = loA.q[1];
    }
    // --- B tile: coeffs are already K-contiguous, just split hi/lo ---
    Pack8 hiB, loB;
    float bv[8] = {bReg0.x, bReg0.y, bReg0.z, bReg0.w,
                   bReg1.x, bReg1.y, bReg1.z, bReg1.w};
#pragma unroll
    for (int j = 0; j < 8; ++j) {
      float rem;
      hiB.s[j] = bf16_split(bv[j], rem);
      loB.s[j] = bf16_rne(rem);
    }
    {
      unsigned short* dH = &sm.Bh[st][bRow * LSTR + bSeg * 8];
      unsigned short* dL = &sm.Bl[st][bRow * LSTR + bSeg * 8];
      *(uint4*)dH = hiB.q;
      *(uint4*)dL = loB.q;
    }
  };

  auto computeS = [&](int st) {
    v16bf aH[2], aL[2], bH[2], bL[2];
    FragU u;
    // A fragment (16x32 bf16): lane holds row M = lane&15.
    // lanes 0-15: K=0..7 and 16..23; lanes 16-31: K=8..15 and 24..31.
#pragma unroll
    for (int m = 0; m < 2; ++m) {
      int row = (wave & 3) * 32 + m * 16 + lr;
      const unsigned short* pH = &sm.Ah[st][row * LSTR + lh * 8];
      const unsigned short* pL = &sm.Al[st][row * LSTR + lh * 8];
      u.q[0] = *(const uint4*)pH;
      u.q[1] = *(const uint4*)(pH + 16);
      aH[m] = u.v;
      u.q[0] = *(const uint4*)pL;
      u.q[1] = *(const uint4*)(pL + 16);
      aL[m] = u.v;
    }
    // B fragment (32x16 bf16): lane holds col N = lane&15.
    // lanes 0-15: K=0..7 and 16..23; lanes 16-31: K=8..15 and 24..31.
#pragma unroll
    for (int n = 0; n < 2; ++n) {
      int col = (wave >> 2) * 32 + n * 16 + lr;
      const unsigned short* pH = &sm.Bh[st][col * LSTR + lh * 8];
      const unsigned short* pL = &sm.Bl[st][col * LSTR + lh * 8];
      u.q[0] = *(const uint4*)pH;
      u.q[1] = *(const uint4*)(pH + 16);
      bH[n] = u.v;
      u.q[0] = *(const uint4*)pL;
      u.q[1] = *(const uint4*)(pL + 16);
      bL[n] = u.v;
    }
    // Error-compensated bf16x2 GEMM: hi*hi + hi*lo + lo*hi, f32 accumulate.
#pragma unroll
    for (int m = 0; m < 2; ++m)
#pragma unroll
      for (int n = 0; n < 2; ++n) {
        acc[m][n] = wmma_bf16(aH[m], bH[n], acc[m][n]);
        acc[m][n] = wmma_bf16(aH[m], bL[n], acc[m][n]);
        acc[m][n] = wmma_bf16(aL[m], bH[n], acc[m][n]);
      }
  };

  // Prologue: fill stage 0.
  loadG(0);
  storeS(0);
  __syncthreads();

  for (int ks = 0; ks < KSTEPS; ++ks) {
    const int cur = ks & 1;
    if (ks + 1 < KSTEPS) loadG(ks + 1);   // issue next global loads early
    computeS(cur);
    if (ks + 1 < KSTEPS) storeS(cur ^ 1); // convert + stage into other buffer
    __syncthreads();
  }

  // Epilogue: C/D layout — VGPR v, lane L -> row v + 8*(L>>4), col L&15.
#pragma unroll
  for (int n = 0; n < 2; ++n) {
    const int col  = oBase + (wave >> 2) * 32 + n * 16 + lr;
    const float bv = bias[col];
#pragma unroll
    for (int m = 0; m < 2; ++m) {
      const int rowBase = mBase + (wave & 3) * 32 + m * 16 + 8 * lh;
#pragma unroll
      for (int v = 0; v < 8; ++v) {
        out[(size_t)(rowBase + v) * OUT_DIMC + col] = acc[m][n][v] + bv;
      }
    }
  }
}

extern "C" void kernel_launch(void* const* d_in, const int* in_sizes, int n_in,
                              void* d_out, int out_size, void* d_ws, size_t ws_size,
                              hipStream_t stream) {
  const float* x      = (const float*)d_in[0];
  const float* coeffs = (const float*)d_in[1];
  const float* bias   = (const float*)d_in[2];
  float* out          = (float*)d_out;

  const int nrows = in_sizes[0] / IN_DIM;   // 8192
  dim3 grid(nrows / BLK_M, OUT_DIMC / BLK_N);
  dim3 block(256);
  taylor_wmma_kernel<<<grid, block, 0, stream>>>(x, coeffs, bias, out);
}
